// GraphSage_14671608283495
// MI455X (gfx1250) — matrix operand order repaired and checked by reading, not compiled, not run on hardware
//
#include <hip/hip_runtime.h>
#include <stdint.h>

// ---------------------------------------------------------------------------
// CDNA5 (gfx1250) GraphSAGE forward, bf16 WMMA path.
// ---------------------------------------------------------------------------

typedef __attribute__((ext_vector_type(16))) __bf16 v16bf;
typedef __attribute__((ext_vector_type(8)))  float  v8f;

union FragU { uint4 u[2]; v16bf v; };

__device__ __forceinline__ uint16_t f32_to_bf16(float f) {
    uint32_t u = __float_as_uint(f);
    u += 0x7FFFu + ((u >> 16) & 1u);          // round-to-nearest-even
    return (uint16_t)(u >> 16);
}

__device__ __forceinline__ v8f wmma_bf16(v16bf a, v16bf b, v8f c) {
    // D = A(16x32 bf16) * B(32x16 bf16) + C(16x16 f32)
    return __builtin_amdgcn_wmma_f32_16x16x32_bf16(
        /*neg_a=*/false, a, /*neg_b=*/false, b,
        /*c_mod=*/(short)0, c, /*reuse_a=*/false, /*reuse_b=*/false);
}

// A-fragment (16x32 bf16, M x K) from an LDS tile laid out [16 rows][128 bf16].
// Contiguous k order (B uses the same): lane<16 holds row=lane, K = ks*32 + {0..15};
//                 lane>=16 holds row=lane-16, K = ks*32 + {16..31}.
__device__ __forceinline__ v16bf load_a_frag_lds(const uint16_t* base, int lane, int ks) {
    const int row = lane & 15;
    const int hi  = lane >> 4;
    const uint16_t* p = base + row * 128 + ks * 32 + hi * 16;
    FragU f;
    f.u[0] = *reinterpret_cast<const uint4*>(p);        // 8 bf16
    f.u[1] = *reinterpret_cast<const uint4*>(p + 8);    // 8 bf16, +8 K
    return f.v;
}

// B-fragment (32x16 bf16, K x N) where B = W^T and W is row-major [E][kcols].
// Lane l holds column N = ecol + (l&15), K = ks*32 + (l>>4)*16 .. +16,
// which is 16 contiguous bf16 of W's row -> two b128 loads.
__device__ __forceinline__ v16bf load_b_frag_g(const uint16_t* __restrict__ W,
                                               int kcols, int ecol, int ks, int lane) {
    const int hi = lane >> 4;
    const uint16_t* p = W + (size_t)(ecol + (lane & 15)) * kcols + ks * 32 + hi * 16;
    FragU f;
    f.u[0] = *reinterpret_cast<const uint4*>(p);
    f.u[1] = *reinterpret_cast<const uint4*>(p + 8);
    return f.v;
}

// fp32 -> bf16 conversion (features + weights, done once).
__global__ void f2bf_kernel(const float* __restrict__ s, uint16_t* __restrict__ d, int n) {
    int i = blockIdx.x * 256 + threadIdx.x;
    if (i < n) d[i] = f32_to_bf16(s[i]);
}

// One SAGE layer. Workgroup = 128 threads (4 waves) = 16 nodes.
// TPN = 16-row neighbor tiles per node (2 for S=25 padded to 32, 1 for S=10 -> 16).
template <int TPN, bool FINAL>
__global__ __launch_bounds__(128) void sage_layer_kernel(
    const uint16_t* __restrict__ feat,   // bf16 [N][128]: gather + self source
    const int*      __restrict__ idx,    // [N][S]
    int S,
    const uint16_t* __restrict__ aggw,   // bf16 [128][128] row-major (e, d)
    const float*    __restrict__ aggb,   // f32  [128]
    const uint16_t* __restrict__ linw,   // bf16 [128][256] row-major (e, d)
    uint16_t*       __restrict__ out_bf, // bf16 [N][128]  (!FINAL)
    float*          __restrict__ out_f32)// f32  [N][128]  (FINAL)
{
    const int tid  = threadIdx.x;
    const int lane = tid & 31;
    const int wave = tid >> 5;
    const int blockBase = blockIdx.x * 16;

    __shared__ __align__(16) uint16_t nbuf[2][16][128]; // double-buffered neighbor tile
    __shared__ __align__(16) uint16_t xbuf[16][128];    // self features (bf16)
    __shared__ __align__(16) uint16_t mbuf[16][128];    // max-pooled aggregator out (bf16)
    __shared__ __align__(16) float    outbuf[16][128];  // lin output (f32) for norm
    __shared__ float rinv_s[16];

    const int grow = tid >> 3;  // gather row 0..15
    const int gch  = tid & 7;   // gather 32B chunk 0..7

    // Aggregator weight B-fragments: resident in VGPRs for the whole workgroup.
    v16bf bW[2][4];
#pragma unroll
    for (int nt = 0; nt < 2; ++nt)
#pragma unroll
        for (int ks = 0; ks < 4; ++ks)
            bW[nt][ks] = load_b_frag_g(aggw, 128, wave * 32 + nt * 16, ks, lane);

    const float bias0 = aggb[wave * 32 + (lane & 15)];
    const float bias1 = aggb[wave * 32 + 16 + (lane & 15)];

    // Prologue: self rows and neighbor tile 0.
    {
        const uint4* src = reinterpret_cast<const uint4*>(
            feat + (size_t)(blockBase + grow) * 128 + gch * 16);
        uint4* dst = reinterpret_cast<uint4*>(&xbuf[grow][gch * 16]);
        dst[0] = src[0]; dst[1] = src[1];
    }
    {
        int s = grow;                      // tile 0 -> mt = 0
        if (s >= S) s = 0;                 // pad with neighbor 0 (max-invariant)
        const int nb = idx[(size_t)blockBase * S + s];
        const uint4* src = reinterpret_cast<const uint4*>(
            feat + (size_t)nb * 128 + gch * 16);
        uint4* dst = reinterpret_cast<uint4*>(&nbuf[0][grow][gch * 16]);
        dst[0] = src[0]; dst[1] = src[1];
    }
    __syncthreads();

    const int NTILES = 16 * TPN;
    float rm0 = -3.0e38f, rm1 = -3.0e38f;

    for (int t = 0; t < NTILES; ++t) {
        // Prefetch next neighbor tile into the other LDS buffer.
        if (t + 1 < NTILES) {
            const int tt = t + 1;
            int s = (tt % TPN) * 16 + grow;
            if (s >= S) s = 0;
            const int nb = idx[(size_t)(blockBase + tt / TPN) * S + s];
            const uint4* src = reinterpret_cast<const uint4*>(
                feat + (size_t)nb * 128 + gch * 16);
            uint4* dst = reinterpret_cast<uint4*>(&nbuf[tt & 1][grow][gch * 16]);
            dst[0] = src[0]; dst[1] = src[1];
        }

        // Aggregator GEMM on current tile: (16 x 128) x (128 x 32 per wave).
        const uint16_t* abase = &nbuf[t & 1][0][0];
        v8f acc0 = {0, 0, 0, 0, 0, 0, 0, 0};
        v8f acc1 = {0, 0, 0, 0, 0, 0, 0, 0};
#pragma unroll
        for (int ks = 0; ks < 4; ++ks) {
            v16bf a = load_a_frag_lds(abase, lane, ks);
            acc0 = wmma_bf16(a, bW[0][ks], acc0);
            acc1 = wmma_bf16(a, bW[1][ks], acc1);
        }
        // bias + ReLU + running max over neighbor rows (M dimension).
#pragma unroll
        for (int j = 0; j < 8; ++j) {
            rm0 = fmaxf(rm0, fmaxf(acc0[j] + bias0, 0.0f));
            rm1 = fmaxf(rm1, fmaxf(acc1[j] + bias1, 0.0f));
        }
        if ((t % TPN) == TPN - 1) {
            // merge row-halves (lanes l and l+16 hold the same column).
            float o0 = fmaxf(rm0, __shfl_xor(rm0, 16, 32));
            float o1 = fmaxf(rm1, __shfl_xor(rm1, 16, 32));
            if (lane < 16) {
                const int k = t / TPN;
                mbuf[k][wave * 32 + lane]      = f32_to_bf16(o0);
                mbuf[k][wave * 32 + 16 + lane] = f32_to_bf16(o1);
            }
            rm0 = -3.0e38f; rm1 = -3.0e38f;
        }
        __syncthreads();
    }

    // lin: out[16 nodes][128] = [x | m] (K=256) @ linw^T, per wave 32 columns.
    v8f lac0 = {0, 0, 0, 0, 0, 0, 0, 0};
    v8f lac1 = {0, 0, 0, 0, 0, 0, 0, 0};
#pragma unroll
    for (int ks = 0; ks < 8; ++ks) {
        const uint16_t* abase = (ks < 4) ? &xbuf[0][0] : &mbuf[0][0];
        v16bf a  = load_a_frag_lds(abase, lane, ks & 3);
        v16bf b0 = load_b_frag_g(linw, 256, wave * 32,      ks, lane);
        v16bf b1 = load_b_frag_g(linw, 256, wave * 32 + 16, ks, lane);
        lac0 = wmma_bf16(a, b0, lac0);
        lac1 = wmma_bf16(a, b1, lac1);
    }
    {
        const int col = wave * 32 + (lane & 15);
        const int mr  = (lane >> 4) * 8;
#pragma unroll
        for (int j = 0; j < 8; ++j) {
            outbuf[mr + j][col]      = lac0[j];
            outbuf[mr + j][col + 16] = lac1[j];
        }
    }
    __syncthreads();

    if (tid < 16) {
        float sum = 0.0f;
        for (int e = 0; e < 128; ++e) { float v = outbuf[tid][e]; sum += v * v; }
        rinv_s[tid] = rsqrtf(sum);
    }
    __syncthreads();

    {
        const float  ri   = rinv_s[grow];
        const size_t node = (size_t)(blockBase + grow);
#pragma unroll
        for (int i = 0; i < 16; ++i) {
            const int e = gch * 16 + i;
            const float v = outbuf[grow][e] * ri;
            if constexpr (FINAL) out_f32[node * 128 + e] = v;
            else                 out_bf [node * 128 + e] = f32_to_bf16(v);
        }
    }
}

// ---------------------------------------------------------------------------
// Host launch
// ---------------------------------------------------------------------------
extern "C" void kernel_launch(void* const* d_in, const int* in_sizes, int n_in,
                              void* d_out, int out_size, void* d_ws, size_t ws_size,
                              hipStream_t stream) {
    (void)in_sizes; (void)n_in; (void)out_size; (void)ws_size;

    const float* features = (const float*)d_in[0];   // [50000][128]
    const int*   idx1     = (const int*)  d_in[1];   // [50000][25]
    const int*   idx2     = (const int*)  d_in[2];   // [50000][10]
    const float* agg1_w   = (const float*)d_in[3];   // [128][128]
    const float* agg1_b   = (const float*)d_in[4];   // [128]
    const float* agg2_w   = (const float*)d_in[5];   // [128][128]
    const float* agg2_b   = (const float*)d_in[6];   // [128]
    const float* lin1_w   = (const float*)d_in[7];   // [128][256]
    const float* lin2_w   = (const float*)d_in[8];   // [128][256]

    constexpr int    N       = 50000;
    constexpr int    D       = 128;
    constexpr size_t FEAT_B  = (size_t)N * D * 2;    // 12.8 MB bf16

    uint8_t*  ws  = (uint8_t*)d_ws;
    uint16_t* fx  = (uint16_t*)(ws);                       // features bf16
    uint16_t* h1  = (uint16_t*)(ws + FEAT_B);              // layer-1 output bf16
    uint16_t* w1b = (uint16_t*)(ws + 2 * FEAT_B);          // 32 KB
    uint16_t* w2b = (uint16_t*)(ws + 2 * FEAT_B + 65536);
    uint16_t* l1b = (uint16_t*)(ws + 2 * FEAT_B + 2 * 65536);
    uint16_t* l2b = (uint16_t*)(ws + 2 * FEAT_B + 3 * 65536);

    // One-shot fp32 -> bf16 conversions.
    f2bf_kernel<<<(N * D + 255) / 256, 256, 0, stream>>>(features, fx, N * D);
    f2bf_kernel<<<64, 256, 0, stream>>>(agg1_w, w1b, 128 * 128);
    f2bf_kernel<<<64, 256, 0, stream>>>(agg2_w, w2b, 128 * 128);
    f2bf_kernel<<<128, 256, 0, stream>>>(lin1_w, l1b, 128 * 256);
    f2bf_kernel<<<128, 256, 0, stream>>>(lin2_w, l2b, 128 * 256);

    // Layer 1: S=25 (2 x 16-row tiles per node), output h1 (bf16).
    sage_layer_kernel<2, false><<<N / 16, 128, 0, stream>>>(
        fx, idx1, 25, w1b, agg1_b, l1b, h1, nullptr);

    // Layer 2: S=10 (1 x 16-row tile per node), output fp32 d_out.
    sage_layer_kernel<1, true><<<N / 16, 128, 0, stream>>>(
        h1, idx2, 10, w2b, agg2_b, l2b, nullptr, (float*)d_out);
}
